// SharedSANIModel_21878563406031
// MI455X (gfx1250) — matrix operand order repaired and checked by reading, not compiled, not run on hardware
//
#include <hip/hip_runtime.h>
#include <hip/hip_bf16.h>

#define NB 4096
#define NA 64
#define NATOMS (NB * NA)
#define AEV 1008
#define KP0 1024
#define NSPEC 7
#define D0 256
#define D1 192
#define D2 160
#define D3 8

typedef __bf16 v16bf __attribute__((ext_vector_type(16)));
typedef float  v8f  __attribute__((ext_vector_type(8)));

// ---------- fragment loaders (contiguous k order: k = 16*(lane/16) + s, same for A and B) ----------
__device__ __forceinline__ v16bf load_frag_A(const __bf16* base, int stride,
                                             int kbase, int lane) {
    int m = lane & 15;
    int bk = (lane >> 4) * 16;
    const __bf16* p0 = base + m * stride + kbase + bk;
    union { v16bf v; uint4 q[2]; } u;
    u.q[0] = *(const uint4*)(p0);        // K bk..bk+7
    u.q[1] = *(const uint4*)(p0 + 8);    // K bk+8..bk+15
    return u.v;
}
__device__ __forceinline__ v16bf load_frag_B(const __bf16* base, int kpad,
                                             int colbase, int kbase, int lane) {
    int n = colbase + (lane & 15);
    int bk = (lane >> 4) * 16;
    const __bf16* p0 = base + (size_t)n * kpad + kbase + bk;
    union { v16bf v; uint4 q[2]; } u;
    u.q[0] = *(const uint4*)(p0);
    u.q[1] = *(const uint4*)(p0 + 8);
    return u.v;
}

__device__ __forceinline__ v8f wmma_bf16(v16bf a, v16bf b, v8f c) {
    return __builtin_amdgcn_wmma_f32_16x16x32_bf16(false, a, false, b,
                                                   (short)0, c, false, false);
}

// C/D 16x16 f32 layout: VGPR v: lanes 0-15 -> M=v, lanes 16-31 -> M=v+8; N=lane&15.
__device__ __forceinline__ void store_celu(v8f c, const float* bias, int colbase,
                                           __bf16* dst, int stride, int lane,
                                           float alpha) {
    int n  = colbase + (lane & 15);
    int mb = (lane >> 4) * 8;
    float bv = bias[n];
#pragma unroll
    for (int v = 0; v < 8; ++v) {
        float x = c[v] + bv;
        float y = x > 0.f ? x : alpha * (__expf(x / alpha) - 1.f);
        dst[(mb + v) * stride + n] = (__bf16)y;
    }
}

// ---------- prep kernels ----------
__global__ void k_init(int* p) {
    if (threadIdx.x < 48) p[threadIdx.x] = 0;
}

__global__ void k_cvtW(const float* __restrict__ W, __bf16* __restrict__ dst,
                       int K, int N, int Kpad, int Npad, int total) {
    for (int i = blockIdx.x * blockDim.x + threadIdx.x; i < total;
         i += gridDim.x * blockDim.x) {
        int s = i / (Npad * Kpad);
        int r = i % (Npad * Kpad);
        int n = r / Kpad, k = r % Kpad;
        float v = (k < K && n < N) ? W[((size_t)s * K + k) * N + n] : 0.f;
        dst[i] = (__bf16)v;
    }
}

__global__ void k_hist(const int* __restrict__ species, int* cnt) {
    for (int i = blockIdx.x * blockDim.x + threadIdx.x; i < NATOMS;
         i += gridDim.x * blockDim.x) {
        int s = species[i];
        if (s >= 0 && s < NSPEC) atomicAdd(&cnt[s], 1);
    }
}

__global__ void k_scan(const int* cnt, int* offs) {
    if (threadIdx.x == 0) {
        int a = 0;
        for (int s = 0; s < NSPEC; ++s) { offs[s] = a; a += cnt[s]; }
        offs[NSPEC] = a;
    }
}

__global__ void k_scatter(const int* __restrict__ species, const int* __restrict__ offs,
                          int* cursor, int* sorted) {
    for (int i = blockIdx.x * blockDim.x + threadIdx.x; i < NATOMS;
         i += gridDim.x * blockDim.x) {
        int s = species[i];
        if (s >= 0 && s < NSPEC) {
            int p = atomicAdd(&cursor[s], 1);
            sorted[offs[s] + p] = i;
        }
    }
}

// ---------- main routed-MLP kernel: 1 block = 64 atoms of one species ----------
// LDS arena (56 KB), lifetime-aliased:
//   [0,32K)  : H1 64x256 bf16; later H3 64x160 at [0,20K)
//   [32K,48K): A-chunk double buffer 2 x (64x64) bf16 (layer0 only)
//   [32K,56K): H2 64x192 bf16
__global__ __launch_bounds__(256) void k_mlp(
    const float* __restrict__ aev,
    const __bf16* __restrict__ wT0, const __bf16* __restrict__ wT1,
    const __bf16* __restrict__ wT2, const __bf16* __restrict__ wT3,
    const float* __restrict__ b0, const float* __restrict__ b1,
    const float* __restrict__ b2, const float* __restrict__ b3,
    const int* __restrict__ cnt, const int* __restrict__ offs,
    const int* __restrict__ sorted, float* __restrict__ atomOut) {

    __shared__ __align__(16) char arena[57344];
    __bf16* sH1 = (__bf16*)(arena);            // 64x256
    __bf16* sAc = (__bf16*)(arena + 32768);    // 2 x 64x64
    __bf16* sH2 = (__bf16*)(arena + 32768);    // 64x192
    __bf16* sH3 = (__bf16*)(arena);            // 64x160
    __shared__ int sAtom[64];
    __shared__ int sSpec, sTileSh;

    const int tid = threadIdx.x, lane = tid & 31, wv = tid >> 5;

    if (tid == 0) {
        int b = blockIdx.x, acc = 0, spec = -1, tile = 0;
        for (int s = 0; s < NSPEC; ++s) {
            int t = (cnt[s] + 63) >> 6;
            if (b < acc + t) { spec = s; tile = b - acc; break; }
            acc += t;
        }
        sSpec = spec; sTileSh = tile;
    }
    __syncthreads();
    const int spec = sSpec;
    if (spec < 0) return;

    if (tid < 64) {
        int pos = sTileSh * 64 + tid;
        sAtom[tid] = (pos < cnt[spec]) ? sorted[offs[spec] + pos] : -1;
    }
    __syncthreads();

    // ---- layer 0: 1008->256 (K padded 1024); M=64, wave owns 2 n-tiles ----
    {
        const __bf16* wb = wT0 + (size_t)spec * D0 * KP0;
        const int cb0 = wv * 32, cb1 = cb0 + 16;
        v8f acc[4][2] = {};
        const int m  = tid >> 2;            // staging row 0..63
        const int kb = (tid & 3) * 16;      // staging col base within chunk
        // Padding rows (atom<0) read a clamped valid row: GEMM rows are
        // independent and padded rows are never stored, so garbage is harmless.
        const int atom = sAtom[m];
        const float* srow = aev + (size_t)(atom < 0 ? 0 : atom) * AEV;

        auto cvt16 = [](float4 f0, float4 f1, float4 f2, float4 f3,
                        __bf16* h) {
            h[0]  = (__bf16)f0.x; h[1]  = (__bf16)f0.y;
            h[2]  = (__bf16)f0.z; h[3]  = (__bf16)f0.w;
            h[4]  = (__bf16)f1.x; h[5]  = (__bf16)f1.y;
            h[6]  = (__bf16)f1.z; h[7]  = (__bf16)f1.w;
            h[8]  = (__bf16)f2.x; h[9]  = (__bf16)f2.y;
            h[10] = (__bf16)f2.z; h[11] = (__bf16)f2.w;
            h[12] = (__bf16)f3.x; h[13] = (__bf16)f3.y;
            h[14] = (__bf16)f3.z; h[15] = (__bf16)f3.w;
        };
        auto compute_chunk = [&](const __bf16* buf, v16bf bA0, v16bf bB0,
                                 v16bf bA1, v16bf bB1) {
#pragma unroll
            for (int mt = 0; mt < 4; ++mt) {
                v16bf a = load_frag_A(buf + mt * 16 * 64, 64, 0, lane);
                acc[mt][0] = wmma_bf16(a, bA0, acc[mt][0]);
                acc[mt][1] = wmma_bf16(a, bB0, acc[mt][1]);
            }
#pragma unroll
            for (int mt = 0; mt < 4; ++mt) {
                v16bf a = load_frag_A(buf + mt * 16 * 64, 64, 32, lane);
                acc[mt][0] = wmma_bf16(a, bA1, acc[mt][0]);
                acc[mt][1] = wmma_bf16(a, bB1, acc[mt][1]);
            }
        };

        // 15 full chunks: branch-free staging (max k = 896+48+15 = 959 < 1008)
        for (int kc = 0; kc < 960; kc += 64) {
            __bf16* buf = sAc + ((kc >> 6) & 1) * (64 * 64);
            union { __bf16 h[16]; uint4 q[2]; } t;
            const float4* p = (const float4*)(srow + kc + kb);
            cvt16(p[0], p[1], p[2], p[3], t.h);
            *(uint4*)(buf + m * 64 + kb)     = t.q[0];
            *(uint4*)(buf + m * 64 + kb + 8) = t.q[1];
            // prefetch this chunk's B panels before the barrier (independent
            // of LDS staging) so their latency hides behind the barrier drain
            v16bf bA0 = load_frag_B(wb, KP0, cb0, kc,      lane);
            v16bf bB0 = load_frag_B(wb, KP0, cb1, kc,      lane);
            v16bf bA1 = load_frag_B(wb, KP0, cb0, kc + 32, lane);
            v16bf bB1 = load_frag_B(wb, KP0, cb1, kc + 32, lane);
            __syncthreads();
            compute_chunk(buf, bA0, bB0, bA1, bB1);
        }
        // peeled final chunk kc=960: k 960..1023, valid up to 1007
        {
            __bf16* buf = sAc + (64 * 64);   // chunk 15 -> buffer 1
            union { __bf16 h[16]; uint4 q[2]; } t;
            if (kb < 48) {
                const float4* p = (const float4*)(srow + 960 + kb);
                cvt16(p[0], p[1], p[2], p[3], t.h);
            } else {
                t.q[0] = uint4{0u, 0u, 0u, 0u};
                t.q[1] = uint4{0u, 0u, 0u, 0u};
            }
            *(uint4*)(buf + m * 64 + kb)     = t.q[0];
            *(uint4*)(buf + m * 64 + kb + 8) = t.q[1];
            v16bf bA0 = load_frag_B(wb, KP0, cb0, 960, lane);
            v16bf bB0 = load_frag_B(wb, KP0, cb1, 960, lane);
            v16bf bA1 = load_frag_B(wb, KP0, cb0, 992, lane);
            v16bf bB1 = load_frag_B(wb, KP0, cb1, 992, lane);
            __syncthreads();
            compute_chunk(buf, bA0, bB0, bA1, bB1);
        }
#pragma unroll
        for (int mt = 0; mt < 4; ++mt) {
            store_celu(acc[mt][0], b0 + spec * D0, cb0, sH1 + mt * 16 * D0, D0, lane, 0.1f);
            store_celu(acc[mt][1], b0 + spec * D0, cb1, sH1 + mt * 16 * D0, D0, lane, 0.1f);
        }
    }
    __syncthreads();

    // ---- layer 1: 256->192; 12 n-tiles over 8 waves ----
    {
        const __bf16* wb = wT1 + (size_t)spec * D1 * D0;
        const int cb0 = wv * 16, cb1 = 128 + wv * 16;
        const bool two = (wv < 4);
        v8f acc[4][2] = {};
#pragma unroll
        for (int kk = 0; kk < D0; kk += 32) {
            v16bf bA = load_frag_B(wb, D0, cb0, kk, lane);
#pragma unroll
            for (int mt = 0; mt < 4; ++mt) {
                v16bf a = load_frag_A(sH1 + mt * 16 * D0, D0, kk, lane);
                acc[mt][0] = wmma_bf16(a, bA, acc[mt][0]);
                if (two) {
                    v16bf bB = load_frag_B(wb, D0, cb1, kk, lane);
                    acc[mt][1] = wmma_bf16(a, bB, acc[mt][1]);
                }
            }
        }
#pragma unroll
        for (int mt = 0; mt < 4; ++mt) {
            store_celu(acc[mt][0], b1 + spec * D1, cb0, sH2 + mt * 16 * D1, D1, lane, 0.1f);
            if (two)
                store_celu(acc[mt][1], b1 + spec * D1, cb1, sH2 + mt * 16 * D1, D1, lane, 0.1f);
        }
    }
    __syncthreads();

    // ---- layer 2: 192->160; 10 n-tiles ----
    {
        const __bf16* wb = wT2 + (size_t)spec * D2 * D1;
        const int cb0 = wv * 16, cb1 = 128 + wv * 16;
        const bool two = (wv < 2);
        v8f acc[4][2] = {};
#pragma unroll
        for (int kk = 0; kk < D1; kk += 32) {
            v16bf bA = load_frag_B(wb, D1, cb0, kk, lane);
#pragma unroll
            for (int mt = 0; mt < 4; ++mt) {
                v16bf a = load_frag_A(sH2 + mt * 16 * D1, D1, kk, lane);
                acc[mt][0] = wmma_bf16(a, bA, acc[mt][0]);
                if (two) {
                    v16bf bB = load_frag_B(wb, D1, cb1, kk, lane);
                    acc[mt][1] = wmma_bf16(a, bB, acc[mt][1]);
                }
            }
        }
#pragma unroll
        for (int mt = 0; mt < 4; ++mt) {
            store_celu(acc[mt][0], b2 + spec * D2, cb0, sH3 + mt * 16 * D2, D2, lane, 0.1f);
            if (two)
                store_celu(acc[mt][1], b2 + spec * D2, cb1, sH3 + mt * 16 * D2, D2, lane, 0.1f);
        }
    }
    __syncthreads();

    // ---- layer 3: 160->8 (N padded to 16); waves 0-3, one m-tile each ----
    if (wv < 4) {
        const __bf16* wb = wT3 + (size_t)spec * 16 * D2;
        v8f c = {};
#pragma unroll
        for (int kk = 0; kk < D2; kk += 32) {
            v16bf a  = load_frag_A(sH3 + wv * 16 * D2, D2, kk, lane);
            v16bf bA = load_frag_B(wb, D2, 0, kk, lane);
            c = wmma_bf16(a, bA, c);
        }
        int n = lane & 15, mb = (lane >> 4) * 8;
        if (n < D3) {
            float bv = b3[spec * D3 + n];
#pragma unroll
            for (int v = 0; v < 8; ++v) {
                int atom = sAtom[wv * 16 + mb + v];
                if (atom >= 0) atomOut[(size_t)atom * D3 + n] = c[v] + bv;
            }
        }
    }
}

// ---------- per-molecule reduction + tiny shared MLP ----------
__global__ __launch_bounds__(64) void k_mol(
    const int* __restrict__ species, const float* __restrict__ coords,
    const float* __restrict__ charge, const float* __restrict__ atomOut,
    const float* __restrict__ Ws0, const float* __restrict__ bs0,
    const float* __restrict__ Ws1, const float* __restrict__ bs1,
    const float* __restrict__ Ws2, const float* __restrict__ bs2,
    float* __restrict__ energ) {

    __shared__ float red[64];
    __shared__ float ssum[8];
    __shared__ float cent[3];
    __shared__ float sfeat[20];

    const int b = blockIdx.x, t = threadIdx.x;
    const int g = b * NA + t;
    const float mval = (species[g] != -1) ? 1.f : 0.f;

    auto rsum = [&](float v) -> float {
        red[t] = v; __syncthreads();
        for (int s = 32; s > 0; s >>= 1) { if (t < s) red[t] += red[t + s]; __syncthreads(); }
        float r = red[0]; __syncthreads();
        return r;
    };

    float num = rsum(mval);
    for (int d = 0; d < D3; ++d) {
        float s = rsum(atomOut[(size_t)g * D3 + d]);
        if (t == 0) ssum[d] = s;
    }
    for (int c = 0; c < 3; ++c) {
        float s = rsum(coords[(size_t)g * 3 + c]);
        if (t == 0) cent[c] = s / num;
    }
    __syncthreads();
    float dx = coords[(size_t)g * 3 + 0] - cent[0];
    float dy = coords[(size_t)g * 3 + 1] - cent[1];
    float dz = coords[(size_t)g * 3 + 2] - cent[2];
    float d2 = (dx * dx + dy * dy + dz * dz) * mval;
    float dist = sqrtf(d2);
    float sum_dist = rsum(dist);
    red[t] = dist; __syncthreads();
    for (int s = 32; s > 0; s >>= 1) { if (t < s) red[t] = fmaxf(red[t], red[t + s]); __syncthreads(); }
    float maxd = red[0]; __syncthreads();
    float se = rsum(__expf(dist - maxd) * mval);
    float smoothmax = __logf(se) + maxd;

    if (t < 8)                sfeat[t] = ssum[t];
    else if (t < 16)          sfeat[t] = ssum[t - 8] / num;
    else if (t == 16)         sfeat[16] = sum_dist;
    else if (t == 17)         sfeat[17] = sum_dist / num;
    else if (t == 18)         sfeat[18] = smoothmax;
    else if (t == 19)         sfeat[19] = charge[b];
    __syncthreads();

    if (t == 0) {
        float h0[32];
        for (int j = 0; j < 32; ++j) {
            float a = bs0[j];
            for (int i = 0; i < 20; ++i) a += sfeat[i] * Ws0[i * 32 + j];
            h0[j] = a > 0.f ? a : (__expf(a) - 1.f);
        }
        float h1[16];
        for (int j = 0; j < 16; ++j) {
            float a = bs1[j];
            for (int i = 0; i < 32; ++i) a += h0[i] * Ws1[i * 16 + j];
            h1[j] = a > 0.f ? a : (__expf(a) - 1.f);
        }
        float o = bs2[0];
        for (int i = 0; i < 16; ++i) o += h1[i] * Ws2[i];
        energ[b] = o;
    }
}

// ---------- host launcher ----------
extern "C" void kernel_launch(void* const* d_in, const int* in_sizes, int n_in,
                              void* d_out, int out_size, void* d_ws, size_t ws_size,
                              hipStream_t stream) {
    (void)in_sizes; (void)n_in; (void)out_size; (void)ws_size;
    const int*   species = (const int*)d_in[0];
    const float* coords  = (const float*)d_in[1];
    const float* charge  = (const float*)d_in[2];
    const float* aev     = (const float*)d_in[3];
    const float* W0 = (const float*)d_in[4],  *b0 = (const float*)d_in[5];
    const float* W1 = (const float*)d_in[6],  *b1 = (const float*)d_in[7];
    const float* W2 = (const float*)d_in[8],  *b2 = (const float*)d_in[9];
    const float* W3 = (const float*)d_in[10], *b3 = (const float*)d_in[11];
    const float* Ws0 = (const float*)d_in[12], *bs0 = (const float*)d_in[13];
    const float* Ws1 = (const float*)d_in[14], *bs1 = (const float*)d_in[15];
    const float* Ws2 = (const float*)d_in[16], *bs2 = (const float*)d_in[17];

    char* ws = (char*)d_ws;
    const size_t SZ_WT0 = (size_t)NSPEC * D0 * KP0 * 2;
    const size_t SZ_WT1 = (size_t)NSPEC * D1 * D0 * 2;
    const size_t SZ_WT2 = (size_t)NSPEC * D2 * D1 * 2;
    const size_t SZ_WT3 = (size_t)NSPEC * 16 * D2 * 2;
    __bf16* wT0 = (__bf16*)(ws);
    __bf16* wT1 = (__bf16*)(ws + SZ_WT0);
    __bf16* wT2 = (__bf16*)(ws + SZ_WT0 + SZ_WT1);
    __bf16* wT3 = (__bf16*)(ws + SZ_WT0 + SZ_WT1 + SZ_WT2);
    char*   pI  = ws + SZ_WT0 + SZ_WT1 + SZ_WT2 + SZ_WT3;
    int* cnt    = (int*)(pI);
    int* offs   = (int*)(pI + 64);
    int* cursor = (int*)(pI + 128);
    int* sorted = (int*)(pI + 192);
    float* atomOut = (float*)(pI + 192 + (size_t)NATOMS * 4);

    k_init<<<1, 64, 0, stream>>>(cnt);

    k_cvtW<<<2048, 256, 0, stream>>>(W0, wT0, AEV, D0, KP0, D0, NSPEC * D0 * KP0);
    k_cvtW<<<1024, 256, 0, stream>>>(W1, wT1, D0, D1, D0, D1, NSPEC * D1 * D0);
    k_cvtW<<<1024, 256, 0, stream>>>(W2, wT2, D1, D2, D1, D2, NSPEC * D2 * D1);
    k_cvtW<<<64,   256, 0, stream>>>(W3, wT3, D2, D3, D2, 16, NSPEC * 16 * D2);

    k_hist<<<512, 256, 0, stream>>>(species, cnt);
    k_scan<<<1, 32, 0, stream>>>(cnt, offs);
    k_scatter<<<512, 256, 0, stream>>>(species, offs, cursor, sorted);

    k_mlp<<<4103, 256, 0, stream>>>(aev, wT0, wT1, wT2, wT3,
                                    b0, b1, b2, b3,
                                    cnt, offs, sorted, atomOut);

    float* energies = (float*)d_out + NATOMS;
    k_mol<<<NB, 64, 0, stream>>>(species, coords, charge, atomOut,
                                 Ws0, bs0, Ws1, bs1, Ws2, bs2, energies);

    hipMemcpyAsync(d_out, d_in[0], (size_t)NATOMS * sizeof(int),
                   hipMemcpyDeviceToDevice, stream);
}
